// T_LSTM_83992380440922
// MI455X (gfx1250) — matrix operand order repaired and checked by reading, not compiled, not run on hardware
//
#include <hip/hip_runtime.h>

// ---------------------------------------------------------------------------
// T-LSTM persistent-scan kernel for gfx1250 (MI455X), bf16 WMMA + f32 accum.
// Batch rows are independent in the recurrence -> each workgroup owns a batch
// tile and keeps h/c state resident across all 255 steps (no grid sync).
//   - c recurrence state: f32, lane-private registers (D-layout stationary)
//   - h/c GEMM operands : bf16 shadows in LDS, stored in A-feed format
//   - WhT B-tiles       : register-resident across the scan (WiT re-read from global per step)
//   - gates             : hardware V_TANH_F32 (sigmoid = 0.5*tanh(z/2)+0.5)
// ---------------------------------------------------------------------------

typedef __attribute__((ext_vector_type(16))) __bf16 v16bf;
typedef __attribute__((ext_vector_type(8)))  __bf16 v8bf;
typedef __attribute__((ext_vector_type(8)))  float  v8f;
typedef __attribute__((ext_vector_type(4)))  float  v4f;

#define BATCH   2048
#define SEQ     256
#define DIN     128
#define HID     256
#define TSTEPS  (SEQ - 1)     // 255 scan steps
#define BM      16            // batch rows per workgroup
#define C2_OFF  2.7193f
#define BHELEMS (BATCH * HID) // one output plane

// ---- WMMA wrapper: D = A(16x32 bf16) * B(32x16 bf16) + C(16x16 f32) -------
__device__ __forceinline__ v8f wmma_bf16(v16bf a, v16bf b, v8f c) {
    return __builtin_amdgcn_wmma_f32_16x16x32_bf16(
        /*neg_a=*/false, a, /*neg_b=*/false, b,
        /*c_mod=*/(short)0, c, /*reuse_a=*/false, /*reuse_b=*/false);
}

// ---- hardware transcendental tanh (CDNA5 V_TANH_F32) -----------------------
// TRANS-op hazard rule: 1 independent op / V_NOP before result use; the
// scheduler cannot see into inline asm, so pad with v_nop explicitly.
__device__ __forceinline__ float htanh(float x) {
    float y;
    asm("v_tanh_f32 %0, %1\n\tv_nop" : "=v"(y) : "v"(x));
    return y;
}
__device__ __forceinline__ float hsigmoid(float z) {
    return fmaf(0.5f, htanh(0.5f * z), 0.5f);   // 1/(1+e^-z)
}

// ---- A-operand from bf16 LDS (16-bit A 16x32 layout, ISA 7.12.2) -----------
// 'p' points at rowBase + k0 + ((lane>>4)<<3); elems 0-7 = p[0..7],
// elems 8-15 = p[16..23]. Two aligned 16B ds_load_b128, no conversion.
__device__ __forceinline__ v16bf load_a_lds(const __bf16* __restrict__ p) {
    v8bf lo = *(const v8bf*)(p);
    v8bf hi = *(const v8bf*)(p + 16);
    return __builtin_shufflevector(lo, hi, 0, 1, 2, 3, 4, 5, 6, 7,
                                           8, 9, 10, 11, 12, 13, 14, 15);
}

// ---- A-operand from f32 global (inputs): load + convert --------------------
__device__ __forceinline__ v16bf load_a_f32g(const float* __restrict__ p) {
    v16bf a;
#pragma unroll
    for (int i = 0; i < 8; ++i) {
        a[i]     = (__bf16)p[i];
        a[8 + i] = (__bf16)p[16 + i];
    }
    return a;
}

// ---- B-operand (32x16 bf16: elems 0-7 = p[0..7], elems 8-15 = p[16..23]; p at k0+8h) --
__device__ __forceinline__ v16bf load_b16(const __bf16* __restrict__ p) {
    v8bf lo = *(const v8bf*)(p);
    v8bf hi = *(const v8bf*)(p + 16);
    return __builtin_shufflevector(lo, hi, 0, 1, 2, 3, 4, 5, 6, 7,
                                           8, 9, 10, 11, 12, 13, 14, 15);
}

// ---------------------------------------------------------------------------
// Prep: transpose + convert weights to bf16 so B-operands are contiguous.
// ---------------------------------------------------------------------------
extern "C" __global__ void __launch_bounds__(256)
prep_weights(const float* __restrict__ Wi, const float* __restrict__ Wh,
             unsigned short* __restrict__ WiT_u, unsigned short* __restrict__ WhT_u) {
    __bf16* WiT = (__bf16*)WiT_u;
    __bf16* WhT = (__bf16*)WhT_u;
    int idx = blockIdx.x * 256 + threadIdx.x;
    if (idx < HID * DIN) {
        int n = idx / DIN, k = idx - n * DIN;
        WiT[idx] = (__bf16)Wi[k * HID + n];
    }
    if (idx < HID * HID) {
        int n = idx >> 8, k = idx & (HID - 1);
        WhT[idx] = (__bf16)Wh[k * HID + n];
    }
}

// ---------------------------------------------------------------------------
// Persistent scan kernel. Grid = BATCH/BM workgroups, 256 threads (8 waves).
// Wave w owns output columns [32w, 32w+32) -> 2 WMMA tiles per path.
// ---------------------------------------------------------------------------
extern "C" __global__ void __launch_bounds__(256)
tlstm_scan(const float* __restrict__ inputs,   // [B][S][DIN]
           const float* __restrict__ timep,    // [B][S]
           const float* __restrict__ bi,       // [HID]
           const float* __restrict__ bh,       // [HID]
           const unsigned short* __restrict__ WiT_u, // bf16 [HID][DIN]
           const unsigned short* __restrict__ WhT_u, // bf16 [HID][HID]
           float* __restrict__ out)            // [3][B][HID]
{
    extern __shared__ char smem_raw[];
    float*  wt_lds = (float*)smem_raw;            // [TSTEPS][BM] decay weights
    __bf16* h_bf   = (__bf16*)(wt_lds + TSTEPS * BM); // [BM][HID] bf16 h
    __bf16* c_bf   = h_bf + BM * HID;                 // [BM][HID] bf16 c
    __bf16* g_bf   = c_bf + BM * HID;                 // [BM][HID] last gate

    const __bf16* WiT = (const __bf16*)WiT_u;
    const __bf16* WhT = (const __bf16*)WhT_u;

    const int tid  = threadIdx.x;
    const int wave = tid >> 5;
    const int lane = tid & 31;
    const int b0   = blockIdx.x * BM;
    const int n0   = wave * 32;                // wave's first output column
    const int lcol = lane & 15;
    const int lhiA = (lane >> 4) << 3;         // A layout K sub-offset (0/8)
    const int lhiB = (lane >> 4) << 3;         // B layout K sub-offset (0/8)

    // zero-init recurrent state shadows
    const __bf16 bz = (__bf16)0.0f;
    for (int i = tid; i < BM * HID; i += 256) { h_bf[i] = bz; c_bf[i] = bz; }

    // precompute all decay weights wt[t][bl] = 1/log(dt + c2)
    for (int i = tid; i < TSTEPS * BM; i += 256) {
        int tt = i >> 4;            // BM == 16
        int bl = i & (BM - 1);
        const float* tp = timep + (size_t)(b0 + bl) * SEQ + tt;
        wt_lds[i] = 1.0f / __logf(tp[1] - tp[0] + C2_OFF);
    }

    // per-lane biases for this lane's two output columns
    float bsum[2], bhn[2];
#pragma unroll
    for (int ni = 0; ni < 2; ++ni) {
        int n = n0 + ni * 16 + lcol;
        bhn[ni]  = bh[n];
        bsum[ni] = bhn[ni] + bi[n];          // x_proj carries +bi, z carries +bh
    }

    // hoist the 16 WhT B-tiles (t-invariant) into registers
    v16bf bWh[8][2];
#pragma unroll
    for (int kk = 0; kk < 8; ++kk)
#pragma unroll
        for (int ni = 0; ni < 2; ++ni)
            bWh[kk][ni] = load_b16(WhT + (n0 + ni * 16 + lcol) * HID + kk * 32 + lhiB);

    __syncthreads();

    const v8f vzero = {0.f, 0.f, 0.f, 0.f, 0.f, 0.f, 0.f, 0.f};
    v8f cReg[2] = {vzero, vzero};              // f32 cell state, lane-private

    const int arow = lcol;                     // A-operand local row
    const float* a_in_row = inputs + (size_t)(b0 + arow) * SEQ * DIN;
    const __bf16* a_h = h_bf + arow * HID + lhiA;
    const __bf16* a_c = c_bf + arow * HID + lhiA;

    for (int t = 0; t < TSTEPS; ++t) {
        v8f accZ[2] = {vzero, vzero};   // x@Wi + h@Wh  (gate pre-activation)
        v8f accS[2] = {vzero, vzero};   // c@Wh         (short-term memory path)

        // ---- x-projection GEMM, K = DIN (f32 global + cvt) --------------
        const float* a_in = a_in_row + (size_t)t * DIN + lhiA;
#pragma unroll
        for (int k = 0; k < DIN; k += 32) {
            v16bf a = load_a_f32g(a_in + k);
#pragma unroll
            for (int ni = 0; ni < 2; ++ni) {
                v16bf b = load_b16(WiT + (n0 + ni * 16 + lcol) * DIN + k + lhiB);
                accZ[ni] = wmma_bf16(a, b, accZ[ni]);
            }
        }

        // ---- recurrent GEMMs, K = HID; A from bf16 LDS, B from registers
#pragma unroll
        for (int kk = 0; kk < 8; ++kk) {
            v16bf ah = load_a_lds(a_h + kk * 32);
            v16bf ac = load_a_lds(a_c + kk * 32);
#pragma unroll
            for (int ni = 0; ni < 2; ++ni) {
                accZ[ni] = wmma_bf16(ah, bWh[kk][ni], accZ[ni]);
                accS[ni] = wmma_bf16(ac, bWh[kk][ni], accS[ni]);
            }
        }

        // prefetch next step's input rows (global_prefetch_b8)
        if (t + 1 < TSTEPS && tid < BM) {
            __builtin_prefetch(inputs + ((size_t)(b0 + tid) * SEQ + (t + 1)) * DIN, 0, 0);
        }

        // decay weights for this lane's 8 rows: two aligned v4f LDS loads
        const v4f* wp = (const v4f*)(wt_lds + t * BM + lhiA);
        v4f w0 = wp[0], w1 = wp[1];

        __syncthreads();   // all LDS reads of h/c done before updates

        // ---- elementwise gate update on the D-matrix register layout ----
        // element r, lane l -> row M = r + (l>=16 ? 8 : 0), col N = l&15
#pragma unroll
        for (int ni = 0; ni < 2; ++ni) {
            int n = n0 + ni * 16 + lcol;
#pragma unroll
            for (int r = 0; r < 8; ++r) {
                int   bl   = r + lhiA;                    // local batch row
                float wt   = (r < 4) ? w0[r & 3] : w1[r & 3];
                float z    = accZ[ni][r] + bsum[ni];
                float zs   = accS[ni][r] + bhn[ni];
                float g    = hsigmoid(z);                 // f==i==o gate
                float Cb   = htanh(z);
                float Cs   = htanh(zs);
                float cold = cReg[ni][r];
                float cnew = g * (fmaf(Cs, wt, cold - Cs) + Cb);
                float hnew = g * htanh(Cb);
                cReg[ni][r] = cnew;
                h_bf[bl * HID + n] = (__bf16)hnew;
                c_bf[bl * HID + n] = (__bf16)cnew;
                if (t == TSTEPS - 1) {
                    g_bf[bl * HID + n] = (__bf16)g;
                    size_t o = (size_t)(b0 + bl) * HID + n;
                    out[(size_t)BHELEMS + o]     = hnew;   // h plane
                    out[(size_t)2 * BHELEMS + o] = cnew;   // c plane
                }
            }
        }
        __syncthreads();   // updates visible before next step's GEMMs
    }

    // ---- epilogue: res = o @ Wh + bh  (o = last-step gate) ---------------
    v8f accR[2] = {vzero, vzero};
    const __bf16* a_g = g_bf + arow * HID + lhiA;
#pragma unroll
    for (int kk = 0; kk < 8; ++kk) {
        v16bf ag = load_a_lds(a_g + kk * 32);
#pragma unroll
        for (int ni = 0; ni < 2; ++ni)
            accR[ni] = wmma_bf16(ag, bWh[kk][ni], accR[ni]);
    }
#pragma unroll
    for (int ni = 0; ni < 2; ++ni) {
        int n = n0 + ni * 16 + lcol;
#pragma unroll
        for (int r = 0; r < 8; ++r) {
            int bl = r + lhiA;
            out[(size_t)(b0 + bl) * HID + n] = accR[ni][r] + bhn[ni];
        }
    }
}

// ---------------------------------------------------------------------------
extern "C" void kernel_launch(void* const* d_in, const int* in_sizes, int n_in,
                              void* d_out, int out_size, void* d_ws, size_t ws_size,
                              hipStream_t stream) {
    (void)in_sizes; (void)n_in; (void)out_size; (void)ws_size;

    const float* inputs = (const float*)d_in[0];  // [2048][256][128]
    const float* timep  = (const float*)d_in[1];  // [2048][256]
    const float* Wi     = (const float*)d_in[2];  // [128][256]
    const float* bi     = (const float*)d_in[3];  // [256]
    const float* Wh     = (const float*)d_in[4];  // [256][256]
    const float* bh     = (const float*)d_in[5];  // [256]
    float* out = (float*)d_out;                   // [3][2048][256]

    // workspace: bf16 transposed weights (WiT: 64 KB, WhT: 128 KB)
    unsigned short* WiT = (unsigned short*)d_ws;
    unsigned short* WhT = WiT + HID * DIN;

    prep_weights<<<(HID * HID) / 256, 256, 0, stream>>>(Wi, Wh, WiT, WhT);

    size_t lds_bytes = (size_t)TSTEPS * BM * sizeof(float)        // wt table
                     + (size_t)3 * BM * HID * sizeof(unsigned short); // h/c/g bf16
    tlstm_scan<<<BATCH / BM, 256, lds_bytes, stream>>>(
        inputs, timep, bi, bh, WiT, WhT, out);
}
